// LFARN_44805098832263
// MI455X (gfx1250) — compile-verified, matrix operand order checked by reading, not run on hardware
//
#include <hip/hip_runtime.h>
#include <hip/hip_bf16.h>

#define GN 100000
#define GE 1600000
#define IN_DIM 64
#define HID_DIM 128
#define OUT_DIM 64

typedef _Float16 v16h __attribute__((ext_vector_type(16)));
typedef float    v8f  __attribute__((ext_vector_type(8)));

// ---------------------------------------------------------------- zero agg
__global__ __launch_bounds__(256) void gnn_zero_kernel(float* __restrict__ agg) {
  int i = blockIdx.x * 256 + threadIdx.x;           // exactly N*IN/4 threads
  ((float4*)agg)[i] = make_float4(0.f, 0.f, 0.f, 0.f);
}

// ------------------------------------------------------- edge scatter-add
// agg is feature-major [IN_DIM][N]; inp is also [IN_DIM][N].
__global__ __launch_bounds__(256) void gnn_scatter_kernel(const float* __restrict__ inp,
                                                          const int* __restrict__ src,
                                                          const int* __restrict__ dst,
                                                          float* __restrict__ agg) {
  int e = blockIdx.x * 256 + threadIdx.x;
  if (e >= GE) return;
  int s = src[e];
  int d = dst[e];
  const float* ip = inp + s;
  float* ap = agg + d;
#pragma unroll 8
  for (int f = 0; f < IN_DIM; ++f) {
    float v = ip[(size_t)f * GN];
    __hip_atomic_fetch_add(ap + (size_t)f * GN, v,
                           __ATOMIC_RELAXED, __HIP_MEMORY_SCOPE_AGENT);
  }
}

// ------------------------------------------------ fused 2-layer MLP (WMMA)
// Block = 128 threads = 4 waves; each wave owns a 16-node M-tile.
// LDS (50432 B):
//   [0     .. 16384)  W1 B-operands, prepacked per-lane v16h (512 x 32B)
//   [16384 .. 32768)  W2 B-operands, prepacked per-lane v16h (512 x 32B)
//   [32768 .. 33280)  b1 (float[128])
//   [33280 .. 33536)  b2 (float[64])
//   [33536 .. 50432)  per-wave 4KB tiles (async fp32 A tile -> f16 h tile),
//                     later reused as outT float[64][66]
__global__ __launch_bounds__(128) void gnn_mlp_kernel(const float* __restrict__ agg,
                                                      const float* __restrict__ W1,
                                                      const float* __restrict__ b1,
                                                      const float* __restrict__ W2,
                                                      const float* __restrict__ b2,
                                                      float* __restrict__ out) {
  __shared__ __attribute__((aligned(32))) unsigned char smem[50432];
  _Float16* w1h = (_Float16*)smem;
  _Float16* w2h = (_Float16*)(smem + 16384);
  float*    b1s = (float*)(smem + 32768);
  float*    b2s = (float*)(smem + 33280);

  const int tid = threadIdx.x;
  const int w   = tid >> 5;
  const int l   = tid & 31;
  const int lm  = l & 15;
  const int g   = l >> 4;

  const int nodeBase = blockIdx.x * 64 + w * 16;

  // ---- 1) async-copy this wave's fp32 A tile [64 k][16 m] into LDS ------
  // N is a multiple of 16 so tiles are all-valid or all-invalid; clamp the
  // invalid ones to a safe source (results discarded at store time).
  {
    int nodeClamp = nodeBase;
    if (nodeClamp > GN - 16) nodeClamp = GN - 16;
    unsigned ldsBase = (unsigned)(uintptr_t)(smem + 33536 + w * 4096);
#pragma unroll
    for (int t = 0; t < 8; ++t) {
      int c = t * 32 + l;                 // 16B chunk id within tile
      int k = c >> 2;                     // feature row
      int q = c & 3;                      // 4-node group within row
      unsigned ldsAddr = ldsBase + (unsigned)c * 16u;
      unsigned gOff    = (unsigned)((k * GN + nodeClamp + q * 4) * 4);
      asm volatile("global_load_async_to_lds_b128 %0, %1, %2"
                   :: "v"(ldsAddr), "v"(gOff), "s"(agg) : "memory");
    }
  }

  // ---- 2) prepack W1/W2 as per-lane WMMA B-operand vectors (f16) --------
#pragma unroll
  for (int it = 0; it < 4; ++it) {
    int V   = tid + it * 128;             // vector id, 512 total
    int kcI = V >> 8;                     // k-chunk (2)
    int jt  = (V >> 5) & 7;               // j-tile (8)
    int ll  = V & 31;
    int gg  = ll >> 4;
    int jj  = jt * 16 + (ll & 15);
    v16h vec;
#pragma unroll
    for (int e = 0; e < 16; ++e) {
      int vv = e >> 1;
      int k0 = kcI * 32 + ((vv & 3) << 1) + ((vv >> 2) << 4) + (gg << 3) + (e & 1);
      vec[e] = (_Float16)W1[jj * IN_DIM + k0];
    }
    *(v16h*)(w1h + (size_t)V * 16) = vec;
  }
#pragma unroll
  for (int it = 0; it < 4; ++it) {
    int V   = tid + it * 128;
    int kcI = V >> 7;                     // k-chunk (4)
    int jt  = (V >> 5) & 3;               // j-tile (4)
    int ll  = V & 31;
    int gg  = ll >> 4;
    int jj  = jt * 16 + (ll & 15);
    v16h vec;
#pragma unroll
    for (int e = 0; e < 16; ++e) {
      int vv = e >> 1;
      int k0 = kcI * 32 + ((vv & 3) << 1) + ((vv >> 2) << 4) + (gg << 3) + (e & 1);
      vec[e] = (_Float16)W2[jj * HID_DIM + k0];
    }
    *(v16h*)(w2h + (size_t)V * 16) = vec;
  }
  if (tid < HID_DIM) b1s[tid] = b1[tid];
  if (tid < OUT_DIM) b2s[tid] = b2[tid];
  __syncthreads();
  asm volatile("s_wait_asynccnt 0" ::: "memory");   // our A tile is in LDS

  // ---- GEMM1: h = relu(agg @ W1^T + b1)  (M=16, N=128, K=64) ------------
  const float* tileF = (const float*)(smem + 33536 + w * 4096);  // [64][16]
  const v16h*  w1v   = (const v16h*)smem;
  const v16h*  w2v   = (const v16h*)(smem + 16384);

  v8f acc1[8] = {};
#pragma unroll
  for (int kcI = 0; kcI < 2; ++kcI) {
    v16h a;
#pragma unroll
    for (int vv = 0; vv < 8; ++vv) {
      int k0 = kcI * 32 + ((vv & 3) << 1) + ((vv >> 2) << 4) + (g << 3);
      a[2 * vv    ] = (_Float16)tileF[k0 * 16 + lm];
      a[2 * vv + 1] = (_Float16)tileF[(k0 + 1) * 16 + lm];
    }
#pragma unroll
    for (int jt = 0; jt < 8; ++jt) {
      v16h bm = w1v[(kcI * 8 + jt) * 32 + l];
      acc1[jt] = __builtin_amdgcn_wmma_f32_16x16x32_f16(
          false, a, false, bm, (short)0, acc1[jt], false, false);
    }
  }

  // ---- bias+relu; write h directly in GEMM2 A-operand order (f16) -------
  _Float16* hTh = (_Float16*)(smem + 33536 + w * 4096);
#pragma unroll
  for (int jt = 0; jt < 8; ++jt) {
    int j = jt * 16 + lm;
#pragma unroll
    for (int r = 0; r < 8; ++r) {
      float hv = acc1[jt][r] + b1s[j];
      hv = hv > 0.f ? hv : 0.f;
      int mr = r + (g << 3);
      int tIdx = ((jt >> 1) * 32 + (lm >> 3) * 16 + mr) * 16
               + ((((lm >> 1) & 3) | ((jt & 1) << 2)) << 1) + (lm & 1);
      hTh[tIdx] = (_Float16)hv;
    }
  }
  asm volatile("s_wait_dscnt 0" ::: "memory");      // wave-local LDS RAW fence

  // ---- GEMM2: out = h @ W2^T + b2  (M=16, N=64, K=128) ------------------
  const v16h* hTv = (const v16h*)hTh;
  v8f acc2[4] = {};
#pragma unroll
  for (int kcI = 0; kcI < 4; ++kcI) {
    v16h a2 = hTv[kcI * 32 + l];
#pragma unroll
    for (int jt = 0; jt < 4; ++jt) {
      v16h bm = w2v[(kcI * 4 + jt) * 32 + l];
      acc2[jt] = __builtin_amdgcn_wmma_f32_16x16x32_f16(
          false, a2, false, bm, (short)0, acc2[jt], false, false);
    }
  }

  __syncthreads();   // all waves done with their tiles before outT reuse

  float* outT = (float*)(smem + 33536);             // [64][66] padded
#pragma unroll
  for (int jt = 0; jt < 4; ++jt) {
    int j = jt * 16 + lm;
#pragma unroll
    for (int r = 0; r < 8; ++r) {
      int mloc = w * 16 + r + (g << 3);
      outT[j * 66 + mloc] = acc2[jt][r] + b2s[j];
    }
  }
  __syncthreads();

  // ---- coalesced transposed store: out[j*N + node] ----------------------
  const int blkBase = blockIdx.x * 64;
#pragma unroll
  for (int it = 0; it < (OUT_DIM * 16) / 128; ++it) {
    int i = tid + it * 128;
    int j = i >> 4;
    int c = (i & 15) << 2;
    int node = blkBase + c;
    if (node + 3 < GN) {
      float4 v4 = make_float4(outT[j * 66 + c], outT[j * 66 + c + 1],
                              outT[j * 66 + c + 2], outT[j * 66 + c + 3]);
      *(float4*)(out + (size_t)j * GN + node) = v4;
    } else {
      for (int q = 0; q < 4; ++q)
        if (node + q < GN) out[(size_t)j * GN + node + q] = outT[j * 66 + c + q];
    }
  }
}

// ----------------------------------------------------------------- launch
extern "C" void kernel_launch(void* const* d_in, const int* in_sizes, int n_in,
                              void* d_out, int out_size, void* d_ws, size_t ws_size,
                              hipStream_t stream) {
  const float* inp = (const float*)d_in[0];   // [64][N]
  const int*   src = (const int*)d_in[1];     // [E]
  const int*   dst = (const int*)d_in[2];     // [E]
  const float* W1  = (const float*)d_in[3];   // [128][64]
  const float* b1  = (const float*)d_in[4];   // [128]
  const float* W2  = (const float*)d_in[5];   // [64][128]
  const float* b2  = (const float*)d_in[6];   // [64]
  float* out = (float*)d_out;                 // [64][N]
  float* agg = (float*)d_ws;                  // [64][N] f32 scratch, 25.6 MB

  gnn_zero_kernel<<<(GN * IN_DIM / 4) / 256, 256, 0, stream>>>(agg);
  gnn_scatter_kernel<<<(GE + 255) / 256, 256, 0, stream>>>(inp, src, dst, agg);
  gnn_mlp_kernel<<<(GN + 63) / 64, 128, 0, stream>>>(agg, W1, b1, W2, b2, out);
}
